// SynthesizerAttention_5798205849810
// MI455X (gfx1250) — matrix operand order repaired and checked by reading, not compiled, not run on hardware
//
#include <hip/hip_runtime.h>
#include <hip/hip_bf16.h>

// Problem constants (match reference)
#define BB   2
#define TT   2048
#define CC   1024
#define HB   16
#define DKK  64
#define NEGV (-1e10f)

typedef __attribute__((ext_vector_type(16))) __bf16 v16bf;
typedef __attribute__((ext_vector_type(8)))  __bf16 bf16x8;
typedef __attribute__((ext_vector_type(8)))  float  v8f;

__device__ __forceinline__ v16bf mkfrag(bf16x8 lo, bf16x8 hi) {
  return __builtin_shufflevector(lo, hi, 0,1,2,3,4,5,6,7,8,9,10,11,12,13,14,15);
}

// A-operand 16x32 bf16 fragment. Lane (l&15) = row M. Per ISA table:
// VGPR0-3 hold K = hf*8 .. hf*8+7, VGPR4-7 hold K = 16+hf*8 .. 16+hf*8+7.
__device__ __forceinline__ v16bf load_a16x32(const __bf16* rowp, int hf) {
  bf16x8 lo = *(const bf16x8*)(rowp + hf * 8);
  bf16x8 hi = *(const bf16x8*)(rowp + 16 + hf * 8);
  return mkfrag(lo, hi);
}

// B-operand 32x16 bf16 fragment. Lane (l&15) = column N; same K order as A:
// VGPR0-3 hold K = hf*8 .. hf*8+7, VGPR4-7 hold K = 16+hf*8 .. 16+hf*8+7.
__device__ __forceinline__ v16bf load_b32x16(const __bf16* colp, int hf) {
  bf16x8 lo = *(const bf16x8*)(colp + hf * 8);
  bf16x8 hi = *(const bf16x8*)(colp + 16 + hf * 8);
  return mkfrag(lo, hi);
}

__device__ __forceinline__ v8f wmma_bf16(v16bf a, v16bf b, v8f c) {
  return __builtin_amdgcn_wmma_f32_16x16x32_bf16(false, a, false, b, (short)0, c,
                                                 false, false);
}

__device__ __forceinline__ bf16x8 cvt8(v8f a) {
  bf16x8 r;
#pragma unroll
  for (int i = 0; i < 8; ++i) r[i] = (__bf16)a[i];
  return r;
}

// ---------------------------------------------------------------------------
// f32 -> bf16 elementwise convert
__global__ void cvt_bf16_kernel(const float* __restrict__ s, __bf16* __restrict__ d,
                                int n) {
  int i = blockIdx.x * blockDim.x + threadIdx.x;
  if (i < n) d[i] = (__bf16)s[i];
}

// w2 (DK=64 x 2048) f32 -> w2t (2048 x 64) bf16
__global__ void w2_transpose_kernel(const float* __restrict__ w2,
                                    __bf16* __restrict__ w2t) {
  int o = blockIdx.x * blockDim.x + threadIdx.x;
  if (o < TT * DKK) {
    int row = o >> 6;  // key index j
    int d = o & 63;    // feature
    w2t[o] = (__bf16)w2[(size_t)d * TT + row];
  }
}

// ---------------------------------------------------------------------------
// Tiled WMMA GEMM: out = act(A(MxK) @ W(NxK)^T + bias).
// One wave -> 16 rows x 64 cols; block = 4 waves (4 consecutive M-tiles).
// MODE 0: f32 row-major (proj path -> d_out)
// MODE 1: relu, bf16 row-major (w1 path)
// MODE 2: bf16 transposed per-head store to vT[(b*C + c)*T + t] (val path)
template <int MODE>
__global__ __launch_bounds__(128) void gemm16_kernel(
    const __bf16* __restrict__ A, const __bf16* __restrict__ W,
    const float* __restrict__ bias, void* __restrict__ outv, int M, int N, int K) {
  const int ln = threadIdx.x & 15;
  const int hf = (threadIdx.x >> 4) & 1;
  const int w  = threadIdx.x >> 5;
  const int m0 = (blockIdx.y * 4 + w) * 16;
  const int n0 = blockIdx.x * 64;
  if (m0 >= M) return;

  v8f acc[4];
#pragma unroll
  for (int nb = 0; nb < 4; ++nb) {
    float bv = bias[n0 + nb * 16 + ln];
#pragma unroll
    for (int r = 0; r < 8; ++r) acc[nb][r] = bv;
  }

  const __bf16* ar = A + (size_t)(m0 + ln) * K;
  const __bf16* wr = W + (size_t)(n0 + ln) * K;

  for (int k0 = 0; k0 < K; k0 += 32) {
    if (k0 + 256 < K) {
      __builtin_prefetch(ar + k0 + 256, 0, 1);  // global_prefetch_b8
      __builtin_prefetch(wr + k0 + 256, 0, 1);
    }
    v16bf af = load_a16x32(ar + k0, hf);
#pragma unroll
    for (int nb = 0; nb < 4; ++nb) {
      v16bf bfr = load_b32x16(wr + (size_t)nb * 16 * K + k0, hf);
      acc[nb] = wmma_bf16(af, bfr, acc[nb]);
    }
  }

  if (MODE == 0) {
    float* out = (float*)outv;
#pragma unroll
    for (int nb = 0; nb < 4; ++nb)
#pragma unroll
      for (int r = 0; r < 8; ++r)
        out[(size_t)(m0 + 8 * hf + r) * N + n0 + nb * 16 + ln] = acc[nb][r];
  } else if (MODE == 1) {
    __bf16* out = (__bf16*)outv;
#pragma unroll
    for (int nb = 0; nb < 4; ++nb)
#pragma unroll
      for (int r = 0; r < 8; ++r)
        out[(size_t)(m0 + 8 * hf + r) * N + n0 + nb * 16 + ln] =
            (__bf16)fmaxf(acc[nb][r], 0.0f);
  } else {
    // vT layout: flat = (b*C + c) * T + t ; per lane: fixed c, 8 consecutive t.
    __bf16* out = (__bf16*)outv;
    int bidx  = m0 / TT;
    int tbase = (m0 % TT) + 8 * hf;
#pragma unroll
    for (int nb = 0; nb < 4; ++nb) {
      int c = n0 + nb * 16 + ln;
      *(bf16x8*)(out + ((size_t)(bidx * CC + c)) * TT + tbase) = cvt8(acc[nb]);
    }
  }
}

// ---------------------------------------------------------------------------
// Flash-style causal attention with learned (query-only) scores.
// rb : relu(w1(x))  (B*T, C) bf16   -> A operand per head slice
// w2t: (2048, 64) bf16 (w2 transposed) -> B operand for scores
// vT : (B, H, DK, T) bf16            -> B operand for P@V
// yb : (B*T, C) bf16 output (pre-proj)
__global__ __launch_bounds__(128) void attn_kernel(
    const __bf16* __restrict__ rb, const __bf16* __restrict__ w2t,
    const float* __restrict__ b2, const __bf16* __restrict__ vT,
    __bf16* __restrict__ yb) {
  __shared__ __align__(16) __bf16 smem[4 * 16 * 40];  // per-wave 16x32 P tile (+pad)

  const int ln = threadIdx.x & 15;
  const int hf = (threadIdx.x >> 4) & 1;
  const int w  = threadIdx.x >> 5;
  const int bh = blockIdx.y;
  const int b  = bh / HB;
  const int h  = bh % HB;
  const int t0 = (blockIdx.x * 4 + w) * 16;

  // Q fragments (16 rows x 64 features = two K=32 blocks), loaded once.
  const __bf16* qp = rb + ((size_t)(b * TT + t0 + ln)) * CC + h * DKK;
  v16bf qa0 = load_a16x32(qp, hf);
  v16bf qa1 = load_a16x32(qp + 32, hf);

  float mrow[8], lsum[8], sc[8];
#pragma unroll
  for (int r = 0; r < 8; ++r) { mrow[r] = -3.0e38f; lsum[r] = 0.0f; }
  v8f O[4];
#pragma unroll
  for (int nb = 0; nb < 4; ++nb)
#pragma unroll
    for (int r = 0; r < 8; ++r) O[nb][r] = 0.0f;

  __bf16* pl = smem + w * 16 * 40;

  for (int j0 = 0; j0 <= t0 + 15; j0 += 32) {
    // ---- scores S (16x32): two 16-wide column tiles, bias from b2 ----
    v8f S[2];
#pragma unroll
    for (int jn = 0; jn < 2; ++jn) {
      int col = j0 + jn * 16 + ln;
      float bv = b2[col];
      v8f s;
#pragma unroll
      for (int r = 0; r < 8; ++r) s[r] = bv;
      const __bf16* wp = w2t + (size_t)col * DKK;
      s = wmma_bf16(qa0, load_b32x16(wp, hf), s);
      s = wmma_bf16(qa1, load_b32x16(wp + 32, hf), s);
      S[jn] = s;
    }

    // ---- causal mask + online softmax (rows M = 8*hf + r, cols N = ln) ----
#pragma unroll
    for (int r = 0; r < 8; ++r) {
      int trow = t0 + 8 * hf + r;
      float s0 = (j0 + ln      <= trow) ? S[0][r] : NEGV;
      float s1 = (j0 + 16 + ln <= trow) ? S[1][r] : NEGV;
      float mn = fmaxf(s0, s1);
#pragma unroll
      for (int mk = 8; mk >= 1; mk >>= 1) mn = fmaxf(mn, __shfl_xor(mn, mk, 32));
      float mi   = fmaxf(mrow[r], mn);
      float scal = __expf(mrow[r] - mi);
      float p0   = __expf(s0 - mi);
      float p1   = __expf(s1 - mi);
      float rs   = p0 + p1;
#pragma unroll
      for (int mk = 8; mk >= 1; mk >>= 1) rs += __shfl_xor(rs, mk, 32);
      lsum[r] = lsum[r] * scal + rs;
      mrow[r] = mi;
      sc[r]   = scal;
      // stage P (D-layout) into LDS for A-layout reload
      pl[(8 * hf + r) * 40 + ln]      = (__bf16)p0;
      pl[(8 * hf + r) * 40 + 16 + ln] = (__bf16)p1;
    }
#pragma unroll
    for (int nb = 0; nb < 4; ++nb)
#pragma unroll
      for (int r = 0; r < 8; ++r) O[nb][r] *= sc[r];

    // all ds stores from this wave must land before the transposed reload
    asm volatile("s_wait_dscnt 0" ::: "memory");

    v16bf pa = load_a16x32(pl + ln * 40, hf);  // ds_load_b128 x2

    // ---- O += P(16x32) @ V(32x64); Vt rows are K-contiguous ----
#pragma unroll
    for (int nb = 0; nb < 4; ++nb) {
      const __bf16* vp = vT + ((size_t)(bh * DKK + nb * 16 + ln)) * TT + j0;
      O[nb] = wmma_bf16(pa, load_b32x16(vp, hf), O[nb]);
    }
  }

  // ---- normalize and emit y (bf16 row-major for proj GEMM) ----
#pragma unroll
  for (int r = 0; r < 8; ++r) {
    float inv = 1.0f / lsum[r];
    int trow = t0 + 8 * hf + r;
    __bf16* yp = yb + ((size_t)(b * TT + trow)) * CC + h * DKK;
#pragma unroll
    for (int nb = 0; nb < 4; ++nb) yp[nb * 16 + ln] = (__bf16)(O[nb][r] * inv);
  }
}

// ---------------------------------------------------------------------------
extern "C" void kernel_launch(void* const* d_in, const int* in_sizes, int n_in,
                              void* d_out, int out_size, void* d_ws, size_t ws_size,
                              hipStream_t stream) {
  (void)in_sizes; (void)n_in; (void)out_size; (void)ws_size;
  const float* x      = (const float*)d_in[0];
  const float* w1_w   = (const float*)d_in[1];
  const float* w1_b   = (const float*)d_in[2];
  const float* w2     = (const float*)d_in[3];
  const float* b2     = (const float*)d_in[4];
  const float* val_w  = (const float*)d_in[5];
  const float* val_b  = (const float*)d_in[6];
  const float* proj_w = (const float*)d_in[7];
  const float* proj_b = (const float*)d_in[8];

  const int M = BB * TT;  // 4096
  char* ws = (char*)d_ws;
  size_t off = 0;
  auto alloc = [&](size_t bytes) {
    void* p = ws + off;
    off += (bytes + 255) & ~(size_t)255;
    return p;
  };
  __bf16* xb    = (__bf16*)alloc((size_t)M * CC * 2);   // x bf16
  __bf16* w1b   = (__bf16*)alloc((size_t)CC * CC * 2);
  __bf16* valb  = (__bf16*)alloc((size_t)CC * CC * 2);
  __bf16* projb = (__bf16*)alloc((size_t)CC * CC * 2);
  __bf16* w2t   = (__bf16*)alloc((size_t)TT * DKK * 2); // w2 transposed
  __bf16* rb    = (__bf16*)alloc((size_t)M * CC * 2);   // relu(w1(x))
  __bf16* vT    = (__bf16*)alloc((size_t)M * CC * 2);   // v transposed per head
  __bf16* yb    = (__bf16*)alloc((size_t)M * CC * 2);   // attention output
  // total ~38.25 MiB of workspace

  int n;
  n = M * CC;
  cvt_bf16_kernel<<<(n + 255) / 256, 256, 0, stream>>>(x, xb, n);
  n = CC * CC;
  cvt_bf16_kernel<<<(n + 255) / 256, 256, 0, stream>>>(w1_w, w1b, n);
  cvt_bf16_kernel<<<(n + 255) / 256, 256, 0, stream>>>(val_w, valb, n);
  cvt_bf16_kernel<<<(n + 255) / 256, 256, 0, stream>>>(proj_w, projb, n);
  n = TT * DKK;
  w2_transpose_kernel<<<(n + 255) / 256, 256, 0, stream>>>(w2, w2t);

  dim3 gg(CC / 64, M / 64);  // (16, 64), 128 threads = 4 waves
  gemm16_kernel<1><<<gg, 128, 0, stream>>>(xb, w1b, w1_b, rb, M, CC, CC);
  gemm16_kernel<2><<<gg, 128, 0, stream>>>(xb, valb, val_b, vT, M, CC, CC);

  attn_kernel<<<dim3(TT / 64, BB * HB), 128, 0, stream>>>(rb, w2t, b2, vT, yb);

  gemm16_kernel<0><<<gg, 128, 0, stream>>>(yb, projb, proj_b, d_out, M, CC, CC);
}
